// SEGINet_19550691131457
// MI455X (gfx1250) — matrix operand order repaired and checked by reading, not compiled, not run on hardware
//
#include <hip/hip_runtime.h>

#define NN 10000
#define NEDG 160000
#define DD 128
#define AA 16
#define NG 16

typedef __attribute__((ext_vector_type(16))) __bf16 v16bf;
typedef __attribute__((ext_vector_type(8)))  float  v8f;
typedef __attribute__((ext_vector_type(4)))  unsigned int v4u;
typedef __attribute__((ext_vector_type(2)))  unsigned int v2u;

__device__ inline float siluf(float v) { return v / (1.f + __expf(-v)); }
__device__ inline float sigmf(float v) { return 1.f / (1.f + __expf(-v)); }

__device__ inline v8f zero_v8f() {
  v8f z;
#pragma unroll
  for (int i = 0; i < 8; ++i) z[i] = 0.f;
  return z;
}

__device__ inline v8f wmma_bf16(v16bf a, v16bf b, v8f c) {
  // D = A(16x32 bf16) x B(32x16 bf16) + C(16x16 f32)
  return __builtin_amdgcn_wmma_f32_16x16x32_bf16(false, a, false, b, (short)0, c,
                                                 false, false);
}

// Pack 4 floats into 4 bf16 (as uint2) for a single ds_store_b64.
union PackBf4 { __bf16 h[4]; v2u u; };
__device__ inline v2u pack4bf(float a, float b, float c, float d) {
  PackBf4 p;
  p.h[0] = (__bf16)a; p.h[1] = (__bf16)b; p.h[2] = (__bf16)c; p.h[3] = (__bf16)d;
  return p.u;
}

// Build an A-fragment (16x32 bf16, M x K) from an LDS tile (row-major, `stride`
// bf16 per row).  Layout used here (contiguous K, as in k_conv_frag): lane m = lane&15; h = lane>>4;
// elems 0..15 -> K = kt*32 + 16h + e
// Both 8-element halves are 16B contiguous and 16B aligned -> 2x ds_load_b128.
union FragU { v4u q[2]; v16bf v; };
__device__ inline v16bf frag_a(const __bf16* s, int lane, int kt, int stride) {
  int m = lane & 15, hh = lane >> 4;
  const __bf16* row = s + m * stride + kt * 32;
  FragU f;
  f.q[0] = *(const v4u*)(row + 16 * hh);
  f.q[1] = *(const v4u*)(row + 16 * hh + 8);
  return f.v;
}

// ---------------------------------------------------------------------------
// Convert a K x 128 f32 weight (row-major) into B-fragment order bf16:
// chunk (nt,kt): 32 lanes x 16 elems; value = W[kt*32 + 16*(lane>>4) + e][nt*16 + (lane&15)]
// ---------------------------------------------------------------------------
__global__ void __launch_bounds__(256) k_conv_frag(const float* __restrict__ src,
                                                   __bf16* __restrict__ dst, int KT) {
  int i = blockIdx.x * blockDim.x + threadIdx.x;
  int total = 8 * KT * 512;
  if (i >= total) return;
  int chunk  = i >> 9;
  int within = i & 511;
  int lane = within >> 4;
  int e    = within & 15;
  int nt = chunk / KT, kt = chunk % KT;
  int n = nt * 16 + (lane & 15);
  int k = kt * 32 + ((lane >> 4) << 4) + e;
  dst[i] = (__bf16)src[k * 128 + n];
}

// ---------------------------------------------------------------------------
// Spherical harmonics per edge + node-attr accumulation
// ---------------------------------------------------------------------------
__global__ void __launch_bounds__(256) k_sh(const float* __restrict__ pos,
                                            const int* __restrict__ eidx, int ne,
                                            float* __restrict__ ea,
                                            float* __restrict__ nasum,
                                            float* __restrict__ cnt) {
  int e = blockIdx.x * blockDim.x + threadIdx.x;
  if (e >= ne) return;
  int s = eidx[e], r = eidx[ne + e];
  float rx = pos[s * 3 + 0] - pos[r * 3 + 0];
  float ry = pos[s * 3 + 1] - pos[r * 3 + 1];
  float rz = pos[s * 3 + 2] - pos[r * 3 + 2];
  float nrm = sqrtf(rx * rx + ry * ry + rz * rz) + 1e-8f;
  float xx = rx / nrm, yy = ry / nrm, zz = rz / nrm;
  float x2 = xx * xx, y2 = yy * yy, z2 = zz * zz;
  float sh[16];
  sh[0] = 1.f;
  sh[1] = 1.7320508f * xx;
  sh[2] = 1.7320508f * yy;
  sh[3] = 1.7320508f * zz;
  sh[4] = 3.8729833f * xx * yy;
  sh[5] = 3.8729833f * yy * zz;
  sh[6] = 1.118034f * (3.f * z2 - 1.f);
  sh[7] = 3.8729833f * xx * zz;
  sh[8] = 1.9364917f * (x2 - y2);
  sh[9] = 2.0916501f * yy * (3.f * x2 - y2);
  sh[10] = 10.2469508f * xx * yy * zz;
  sh[11] = 1.6201852f * yy * (5.f * z2 - 1.f);
  sh[12] = 1.3228757f * (5.f * z2 - 3.f) * zz;
  sh[13] = 1.6201852f * xx * (5.f * z2 - 1.f);
  sh[14] = 5.1234754f * zz * (x2 - y2);
  sh[15] = 2.0916501f * xx * (x2 - 3.f * y2);
#pragma unroll
  for (int a = 0; a < 16; ++a) {
    ea[(size_t)e * 16 + a] = sh[a];
    atomicAdd(&nasum[(size_t)r * 16 + a], sh[a]);
  }
  atomicAdd(&cnt[r], 1.f);
}

__global__ void __launch_bounds__(256) k_na_final(float* __restrict__ na,
                                                  const float* __restrict__ cnt, int n) {
  int i = blockIdx.x * blockDim.x + threadIdx.x;
  if (i >= n) return;
  float c = cnt[i];
  if (c < 1.f) c = 1.f;
  float inv = 1.f / c;
#pragma unroll
  for (int a = 15; a >= 1; --a) na[(size_t)i * 16 + a] *= inv;
  na[(size_t)i * 16] = 1.f;
}

// ---------------------------------------------------------------------------
// Input embedding: h = silu((x @ Wx + b) * (na @ Wa))  (K = 16, plain VALU)
// ---------------------------------------------------------------------------
__global__ void __launch_bounds__(256) k_emb_in(const float* __restrict__ x,
                                                const float* __restrict__ na,
                                                const float* __restrict__ wx,
                                                const float* __restrict__ wa,
                                                const float* __restrict__ bias,
                                                float* __restrict__ h, int n) {
  int i = blockIdx.x * blockDim.x + threadIdx.x;
  if (i >= n) return;
  float xr[16], nr[16];
#pragma unroll
  for (int k = 0; k < 16; ++k) {
    xr[k] = x[(size_t)i * 16 + k];
    nr[k] = na[(size_t)i * 16 + k];
  }
  for (int f = 0; f < 128; ++f) {
    float v = bias[f], g = 0.f;
#pragma unroll
    for (int k = 0; k < 16; ++k) {
      v += xr[k] * wx[k * 128 + f];
      g += nr[k] * wa[k * 128 + f];
    }
    h[(size_t)i * 128 + f] = siluf(v * g);
  }
}

// ---------------------------------------------------------------------------
// Edge message kernel (WMMA).  One wave == one 16-edge tile.
//   m1 = silu((x[recv]@W1a + x[send]@W1b + dist*w1d + b1) * (ea@Wa1))
//   m2 = silu((m1@W2 + b2) * (ea@Wa2))  -> stored + BN statistics
// ---------------------------------------------------------------------------
__global__ void __launch_bounds__(128) k_edge_mm(
    const float* __restrict__ hin, const int* __restrict__ eidx, int ne,
    const float* __restrict__ dist, const float* __restrict__ ea,
    const v16bf* __restrict__ w1a, const v16bf* __restrict__ w1b,
    const v16bf* __restrict__ w2f, const float* __restrict__ wa1,
    const float* __restrict__ wa2, const float* __restrict__ b1,
    const float* __restrict__ b2, const float* __restrict__ w1d,
    float* __restrict__ m2out, float* __restrict__ bnsum,
    float* __restrict__ bnsq, int n_tiles) {
  extern __shared__ char smem[];
  float* s_bn = (float*)smem;        // 128
  float* s_bq = s_bn + 128;          // 128
  int tid = threadIdx.x, wave = tid >> 5, lane = tid & 31;
  char* wb = smem + 1024 + wave * 13440;
  __bf16* s_xa = (__bf16*)wb;        // 16x128 bf16
  __bf16* s_xb = s_xa + 2048;        // 16x128 bf16
  __bf16* s_m1 = s_xb + 2048;        // 16x128 bf16
  float* s_ea = (float*)(wb + 12288);  // 16x16 f32
  float* s_ds = s_ea + 256;            // 16 f32

  s_bn[tid] = 0.f;
  s_bq[tid] = 0.f;
  __syncthreads();

  int tile = blockIdx.x * 4 + wave;
  if (tile < n_tiles) {
    int e0 = tile * 16;
    for (int i = lane; i < 256; i += 32) s_ea[i] = ea[(size_t)e0 * 16 + i];
    if (lane < 16) s_ds[lane] = dist[e0 + lane];

    // gather x[recv], x[send] -> bf16 LDS tiles (2 lanes per row, 64 f32 each)
    int row = lane >> 1, half = lane & 1;
    int er = eidx[ne + e0 + row];
    int es = eidx[e0 + row];
    const float4* hr = (const float4*)(hin + (size_t)er * 128 + half * 64);
    const float4* hs = (const float4*)(hin + (size_t)es * 128 + half * 64);
    v2u* da = (v2u*)(s_xa + row * 128 + half * 64);
    v2u* db = (v2u*)(s_xb + row * 128 + half * 64);
#pragma unroll
    for (int i = 0; i < 16; ++i) {
      float4 v = hr[i];
      da[i] = pack4bf(v.x, v.y, v.z, v.w);
      float4 u = hs[i];
      db[i] = pack4bf(u.x, u.y, u.z, u.w);
    }

    // ---- matmul 1: [xr|xs] @ [W1a;W1b] ----
    v8f acc[8];
#pragma unroll
    for (int nt = 0; nt < 8; ++nt) acc[nt] = zero_v8f();
#pragma unroll
    for (int kt = 0; kt < 4; ++kt) {
      v16bf bw[8];
#pragma unroll
      for (int nt = 0; nt < 8; ++nt) bw[nt] = w1a[(nt * 4 + kt) * 32 + lane];
      v16bf a = frag_a(s_xa, lane, kt, 128);
#pragma unroll
      for (int nt = 0; nt < 8; ++nt) acc[nt] = wmma_bf16(a, bw[nt], acc[nt]);
    }
#pragma unroll
    for (int kt = 0; kt < 4; ++kt) {
      v16bf bw[8];
#pragma unroll
      for (int nt = 0; nt < 8; ++nt) bw[nt] = w1b[(nt * 4 + kt) * 32 + lane];
      v16bf a = frag_a(s_xb, lane, kt, 128);
#pragma unroll
      for (int nt = 0; nt < 8; ++nt) acc[nt] = wmma_bf16(a, bw[nt], acc[nt]);
    }

    int ncl = lane & 15, mh = (lane >> 4) * 8;
#pragma unroll
    for (int nt = 0; nt < 8; ++nt) {
#pragma unroll
      for (int r = 0; r < 8; ++r) {
        int n = nt * 16 + ncl, m = mh + r;
        float v = acc[nt][r] + b1[n] + s_ds[m] * w1d[n];
        float g = 0.f;
#pragma unroll
        for (int a = 0; a < 16; ++a) g += s_ea[m * 16 + a] * wa1[a * 128 + n];
        s_m1[m * 128 + n] = (__bf16)siluf(v * g);
      }
    }

    // ---- matmul 2: m1 @ W2 ----
    v8f ac2[8];
#pragma unroll
    for (int nt = 0; nt < 8; ++nt) ac2[nt] = zero_v8f();
#pragma unroll
    for (int kt = 0; kt < 4; ++kt) {
      v16bf bw[8];
#pragma unroll
      for (int nt = 0; nt < 8; ++nt) bw[nt] = w2f[(nt * 4 + kt) * 32 + lane];
      v16bf a = frag_a(s_m1, lane, kt, 128);
#pragma unroll
      for (int nt = 0; nt < 8; ++nt) ac2[nt] = wmma_bf16(a, bw[nt], ac2[nt]);
    }

#pragma unroll
    for (int nt = 0; nt < 8; ++nt) {
      float cs = 0.f, cq = 0.f;
#pragma unroll
      for (int r = 0; r < 8; ++r) {
        int n = nt * 16 + ncl, m = mh + r;
        float v = ac2[nt][r] + b2[n];
        float g = 0.f;
#pragma unroll
        for (int a = 0; a < 16; ++a) g += s_ea[m * 16 + a] * wa2[a * 128 + n];
        float sv = siluf(v * g);
        m2out[(size_t)(e0 + m) * 128 + n] = sv;
        cs += sv;
        cq += sv * sv;
      }
      atomicAdd(&s_bn[nt * 16 + ncl], cs);
      atomicAdd(&s_bq[nt * 16 + ncl], cq);
    }
  }
  __syncthreads();
  atomicAdd(&bnsum[tid], s_bn[tid]);
  atomicAdd(&bnsq[tid], s_bq[tid]);
}

__global__ void k_bn_final(const float* __restrict__ bns, const float* __restrict__ bnq,
                           const float* __restrict__ g, const float* __restrict__ b,
                           float* __restrict__ scale, float* __restrict__ shift,
                           float inv_ne) {
  int f = threadIdx.x;
  float mu = bns[f] * inv_ne;
  float var = bnq[f] * inv_ne - mu * mu;
  float sc = g[f] * rsqrtf(var + 1e-5f);
  scale[f] = sc;
  shift[f] = b[f] - mu * sc;
}

// att = sigmoid((bn(m)@iWx + ib) * (ea@iWa)); agg[recv] += bn(m)*att
__global__ void __launch_bounds__(256) k_att_agg(
    const float* __restrict__ m2, const float* __restrict__ ea,
    const float* __restrict__ scale, const float* __restrict__ shift,
    const float* __restrict__ iwx, const float* __restrict__ iwa,
    const float* __restrict__ ib, const int* __restrict__ eidx, int ne,
    float* __restrict__ agg) {
  int e = blockIdx.x * blockDim.x + threadIdx.x;
  if (e >= ne) return;
  const float* mr = m2 + (size_t)e * 128;
  float d = 0.f;
  for (int f = 0; f < 128; ++f) d += (mr[f] * scale[f] + shift[f]) * iwx[f];
  float ga = 0.f;
#pragma unroll
  for (int a = 0; a < 16; ++a) ga += ea[(size_t)e * 16 + a] * iwa[a];
  float att = sigmf((d + ib[0]) * ga);
  int rv = eidx[ne + e];
  float* ao = agg + (size_t)rv * 128;
  for (int f = 0; f < 128; ++f)
    atomicAdd(&ao[f], (mr[f] * scale[f] + shift[f]) * att);
}

// A1 = softmax(agg, axis=1) * h
__global__ void __launch_bounds__(256) k_softmax_prep(const float* __restrict__ agg,
                                                      const float* __restrict__ hin,
                                                      float* __restrict__ A1, int n) {
  int i = blockIdx.x * blockDim.x + threadIdx.x;
  if (i >= n) return;
  const float* ar = agg + (size_t)i * 128;
  float mx = -3.4e38f;
  for (int f = 0; f < 128; ++f) mx = fmaxf(mx, ar[f]);
  float s = 0.f;
  for (int f = 0; f < 128; ++f) s += __expf(ar[f] - mx);
  float inv = 1.f / s;
  const float* hr = hin + (size_t)i * 128;
  float* o = A1 + (size_t)i * 128;
  for (int f = 0; f < 128; ++f) o[f] = __expf(ar[f] - mx) * inv * hr[f];
}

// ---------------------------------------------------------------------------
// Generic node transform (WMMA): out = act(([A1|A2] @ W + b) * (na @ Wa))
// A2 == nullptr -> K = 128, else K = 256.  One wave per 16-node tile.
// ---------------------------------------------------------------------------
__global__ void __launch_bounds__(128) k_node_mm(
    const float* __restrict__ A1, const float* __restrict__ A2,
    const v16bf* __restrict__ wf, const float* __restrict__ wa,
    const float* __restrict__ bias, const float* __restrict__ na,
    float* __restrict__ outp, int n_tiles, int do_silu) {
  extern __shared__ char smem[];
  int tid = threadIdx.x, wave = tid >> 5, lane = tid & 31;
  char* wb = smem + wave * 9216;
  __bf16* s_a = (__bf16*)wb;          // up to 16x256 bf16
  float* s_na = (float*)(wb + 8192);  // 16x16 f32
  int tile = blockIdx.x * 4 + wave;
  if (tile >= n_tiles) return;
  int e0 = tile * 16;
  int KT = A2 ? 8 : 4;
  int K = KT * 32;

  for (int i = lane; i < 256; i += 32) s_na[i] = na[(size_t)e0 * 16 + i];
  int row = lane >> 1, half = lane & 1;
  {
    const float4* p = (const float4*)(A1 + (size_t)(e0 + row) * 128 + half * 64);
    v2u* d = (v2u*)(s_a + row * K + half * 64);
#pragma unroll
    for (int i = 0; i < 16; ++i) {
      float4 v = p[i];
      d[i] = pack4bf(v.x, v.y, v.z, v.w);
    }
  }
  if (A2) {
    const float4* p = (const float4*)(A2 + (size_t)(e0 + row) * 128 + half * 64);
    v2u* d = (v2u*)(s_a + row * K + 128 + half * 64);
#pragma unroll
    for (int i = 0; i < 16; ++i) {
      float4 v = p[i];
      d[i] = pack4bf(v.x, v.y, v.z, v.w);
    }
  }

  v8f acc[8];
#pragma unroll
  for (int nt = 0; nt < 8; ++nt) acc[nt] = zero_v8f();
  for (int kt = 0; kt < KT; ++kt) {
    v16bf bw[8];
#pragma unroll
    for (int nt = 0; nt < 8; ++nt) bw[nt] = wf[(nt * KT + kt) * 32 + lane];
    v16bf a = frag_a(s_a, lane, kt, K);
#pragma unroll
    for (int nt = 0; nt < 8; ++nt) acc[nt] = wmma_bf16(a, bw[nt], acc[nt]);
  }

  int ncl = lane & 15, mh = (lane >> 4) * 8;
#pragma unroll
  for (int nt = 0; nt < 8; ++nt) {
#pragma unroll
    for (int r = 0; r < 8; ++r) {
      int n = nt * 16 + ncl, m = mh + r;
      float v = acc[nt][r] + bias[n];
      float g = 0.f;
#pragma unroll
      for (int a = 0; a < 16; ++a) g += s_na[m * 16 + a] * wa[a * 128 + n];
      float t = v * g;
      if (do_silu) t = siluf(t);
      outp[(size_t)(e0 + m) * 128 + n] = t;
    }
  }
}

__global__ void __launch_bounds__(256) k_pool(const float* __restrict__ hp,
                                              const int* __restrict__ batch,
                                              float* __restrict__ g, int n) {
  int i = blockIdx.x * blockDim.x + threadIdx.x;
  if (i >= n) return;
  int b = batch[i];
  const float* hr = hp + (size_t)i * 128;
  float* go = g + (size_t)b * 128;
  for (int f = 0; f < 128; ++f) atomicAdd(&go[f], hr[f]);
}

__global__ void __launch_bounds__(128) k_head(const float* __restrict__ g,
                                              const float* __restrict__ pp1,
                                              const float* __restrict__ pp1b,
                                              const float* __restrict__ pp2,
                                              const float* __restrict__ pp2b,
                                              float* __restrict__ out) {
  __shared__ float s_g1[16 * 128];
  int f = threadIdx.x;
  for (int gi = 0; gi < 16; ++gi) {
    float acc = pp1b[f];
    for (int k = 0; k < 128; ++k) acc += g[gi * 128 + k] * pp1[k * 128 + f];
    s_g1[gi * 128 + f] = siluf(acc);
  }
  __syncthreads();
  if (f < 16) {
    float o = pp2b[0];
    for (int k = 0; k < 128; ++k) o += s_g1[f * 128 + k] * pp2[k];
    out[f] = o;
  }
}

// ---------------------------------------------------------------------------
extern "C" void kernel_launch(void* const* d_in, const int* in_sizes, int n_in,
                              void* d_out, int out_size, void* d_ws, size_t ws_size,
                              hipStream_t stream) {
  (void)in_sizes; (void)n_in; (void)out_size; (void)ws_size;
  const float* x          = (const float*)d_in[0];
  const float* pos        = (const float*)d_in[1];
  const int*   batch      = (const int*)d_in[2];
  const int*   iei        = (const int*)d_in[3];
  const float* iea        = (const float*)d_in[4];
  const int*   eei        = (const int*)d_in[5];
  const float* eea        = (const float*)d_in[6];
  const float* emb_in_Wx  = (const float*)d_in[7];
  const float* emb_in_Wa  = (const float*)d_in[8];
  const float* emb_in_b   = (const float*)d_in[9];
  const float* emb_out_Wx = (const float*)d_in[10];
  const float* emb_out_Wa = (const float*)d_in[11];
  const float* emb_out_b  = (const float*)d_in[12];
  const float* msg1_Wx    = (const float*)d_in[13];
  const float* msg1_Wa    = (const float*)d_in[14];
  const float* msg1_b     = (const float*)d_in[15];
  const float* msg2_Wx    = (const float*)d_in[16];
  const float* msg2_Wa    = (const float*)d_in[17];
  const float* msg2_b     = (const float*)d_in[18];
  const float* bn_g       = (const float*)d_in[19];
  const float* bn_b       = (const float*)d_in[20];
  const float* inf_Wx     = (const float*)d_in[21];
  const float* inf_Wa     = (const float*)d_in[22];
  const float* inf_b      = (const float*)d_in[23];
  const float* upd_Wx     = (const float*)d_in[24];
  const float* upd_Wa     = (const float*)d_in[25];
  const float* upd_b      = (const float*)d_in[26];
  const float* prepool_Wx = (const float*)d_in[27];
  const float* prepool_Wa = (const float*)d_in[28];
  const float* prepool_b  = (const float*)d_in[29];
  const float* pp1_W      = (const float*)d_in[30];
  const float* pp1_b      = (const float*)d_in[31];
  const float* pp2_W      = (const float*)d_in[32];
  const float* pp2_b      = (const float*)d_in[33];

  char* wsp = (char*)d_ws;
  size_t off = 0;
  auto alloc = [&](size_t bytes) {
    size_t o = off;
    off = (off + bytes + 255) & ~(size_t)255;
    return o;
  };
  size_t o_w1a = alloc(6 * 32768);
  size_t o_w1b = alloc(6 * 32768);
  size_t o_w2  = alloc(6 * 32768);
  size_t o_wu  = alloc(6 * 65536);
  size_t o_weo = alloc(32768);
  size_t o_wpp = alloc(32768);
  size_t o_eai = alloc((size_t)NEDG * 16 * 4);
  size_t o_eae = alloc((size_t)NEDG * 16 * 4);
  size_t o_nai = alloc((size_t)NN * 16 * 4);
  size_t o_nae = alloc((size_t)NN * 16 * 4);
  size_t o_cnt = alloc((size_t)NN * 4);
  size_t o_h0  = alloc((size_t)NN * 128 * 4);
  size_t o_h1  = alloc((size_t)NN * 128 * 4);
  size_t o_a1  = alloc((size_t)NN * 128 * 4);
  size_t o_m2  = alloc((size_t)NEDG * 128 * 4);
  size_t o_agg = alloc((size_t)NN * 128 * 4);
  size_t o_bns = alloc(512);
  size_t o_bnq = alloc(512);
  size_t o_bsc = alloc(512);
  size_t o_bsh = alloc(512);
  size_t o_g   = alloc(16 * 128 * 4);

  // ---- weight conversion into B-fragment order (bf16) ----
  for (int l = 0; l < 6; ++l) {
    k_conv_frag<<<64, 256, 0, stream>>>(msg1_Wx + (size_t)l * 257 * 128,
                                        (__bf16*)(wsp + o_w1a) + (size_t)l * 16384, 4);
    k_conv_frag<<<64, 256, 0, stream>>>(msg1_Wx + (size_t)l * 257 * 128 + 128 * 128,
                                        (__bf16*)(wsp + o_w1b) + (size_t)l * 16384, 4);
    k_conv_frag<<<64, 256, 0, stream>>>(msg2_Wx + (size_t)l * 128 * 128,
                                        (__bf16*)(wsp + o_w2) + (size_t)l * 16384, 4);
    k_conv_frag<<<128, 256, 0, stream>>>(upd_Wx + (size_t)l * 256 * 128,
                                         (__bf16*)(wsp + o_wu) + (size_t)l * 32768, 8);
  }
  k_conv_frag<<<64, 256, 0, stream>>>(emb_out_Wx, (__bf16*)(wsp + o_weo), 4);
  k_conv_frag<<<64, 256, 0, stream>>>(prepool_Wx, (__bf16*)(wsp + o_wpp), 4);

  // ---- spherical harmonics + node attrs (internal then external) ----
  hipMemsetAsync(wsp + o_nai, 0, (size_t)NN * 16 * 4, stream);
  hipMemsetAsync(wsp + o_cnt, 0, (size_t)NN * 4, stream);
  k_sh<<<625, 256, 0, stream>>>(pos, iei, NEDG, (float*)(wsp + o_eai),
                                (float*)(wsp + o_nai), (float*)(wsp + o_cnt));
  k_na_final<<<40, 256, 0, stream>>>((float*)(wsp + o_nai), (float*)(wsp + o_cnt), NN);

  hipMemsetAsync(wsp + o_nae, 0, (size_t)NN * 16 * 4, stream);
  hipMemsetAsync(wsp + o_cnt, 0, (size_t)NN * 4, stream);
  k_sh<<<625, 256, 0, stream>>>(pos, eei, NEDG, (float*)(wsp + o_eae),
                                (float*)(wsp + o_nae), (float*)(wsp + o_cnt));
  k_na_final<<<40, 256, 0, stream>>>((float*)(wsp + o_nae), (float*)(wsp + o_cnt), NN);

  // ---- input embedding ----
  k_emb_in<<<40, 256, 0, stream>>>(x, (float*)(wsp + o_nai), emb_in_Wx, emb_in_Wa,
                                   emb_in_b, (float*)(wsp + o_h0), NN);

  float* hc = (float*)(wsp + o_h0);
  float* hn = (float*)(wsp + o_h1);

  auto run_layer = [&](const int* eidx, const float* dist, const float* eabuf,
                       const float* nabuf, int l) {
    hipMemsetAsync(wsp + o_bns, 0, 512, stream);
    hipMemsetAsync(wsp + o_bnq, 0, 512, stream);
    k_edge_mm<<<2500, 128, 54784, stream>>>(
        hc, eidx, NEDG, dist, eabuf,
        (const v16bf*)(wsp + o_w1a) + (size_t)l * 1024,
        (const v16bf*)(wsp + o_w1b) + (size_t)l * 1024,
        (const v16bf*)(wsp + o_w2) + (size_t)l * 1024,
        msg1_Wa + (size_t)l * 2048, msg2_Wa + (size_t)l * 2048,
        msg1_b + (size_t)l * 128, msg2_b + (size_t)l * 128,
        msg1_Wx + (size_t)l * 257 * 128 + 256 * 128,
        (float*)(wsp + o_m2), (float*)(wsp + o_bns), (float*)(wsp + o_bnq), 10000);
    k_bn_final<<<1, 128, 0, stream>>>((float*)(wsp + o_bns), (float*)(wsp + o_bnq),
                                      bn_g + (size_t)l * 128, bn_b + (size_t)l * 128,
                                      (float*)(wsp + o_bsc), (float*)(wsp + o_bsh),
                                      1.0f / (float)NEDG);
    hipMemsetAsync(wsp + o_agg, 0, (size_t)NN * 128 * 4, stream);
    k_att_agg<<<625, 256, 0, stream>>>(
        (float*)(wsp + o_m2), eabuf, (float*)(wsp + o_bsc), (float*)(wsp + o_bsh),
        inf_Wx + (size_t)l * 128, inf_Wa + (size_t)l * 16, inf_b + l, eidx, NEDG,
        (float*)(wsp + o_agg));
    k_softmax_prep<<<40, 256, 0, stream>>>((float*)(wsp + o_agg), hc,
                                           (float*)(wsp + o_a1), NN);
    k_node_mm<<<157, 128, 36864, stream>>>(
        (float*)(wsp + o_a1), (float*)(wsp + o_agg),
        (const v16bf*)(wsp + o_wu) + (size_t)l * 2048, upd_Wa + (size_t)l * 2048,
        upd_b + (size_t)l * 128, nabuf, hn, 625, 1);
    float* t = hc; hc = hn; hn = t;
  };

  for (int l = 0; l < 3; ++l)
    run_layer(iei, iea, (float*)(wsp + o_eai), (float*)(wsp + o_nai), l);

  // emb_out (no silu), gated by external node attrs
  k_node_mm<<<157, 128, 36864, stream>>>(hc, (const float*)nullptr,
                                         (const v16bf*)(wsp + o_weo), emb_out_Wa,
                                         emb_out_b, (float*)(wsp + o_nae), hn, 625, 0);
  { float* t = hc; hc = hn; hn = t; }

  for (int l = 3; l < 6; ++l)
    run_layer(eei, eea, (float*)(wsp + o_eae), (float*)(wsp + o_nae), l);

  // prepool (silu)
  k_node_mm<<<157, 128, 36864, stream>>>(hc, (const float*)nullptr,
                                         (const v16bf*)(wsp + o_wpp), prepool_Wa,
                                         prepool_b, (float*)(wsp + o_nae), hn, 625, 1);
  { float* t = hc; hc = hn; hn = t; }

  // pooling + head
  hipMemsetAsync(wsp + o_g, 0, 16 * 128 * 4, stream);
  k_pool<<<40, 256, 0, stream>>>(hc, batch, (float*)(wsp + o_g), NN);
  k_head<<<1, 128, 0, stream>>>((float*)(wsp + o_g), pp1_W, pp1_b, pp2_W, pp2_b,
                                (float*)d_out);
}
